// RNN_Hierarchical_11330123727576
// MI455X (gfx1250) — compile-verified, matrix operand order checked by reading, not run on hardware
//
#include <hip/hip_runtime.h>

// CDNA5 (gfx1250) persistent-RNN kernel.
// Grid = 32 blocks x 256 threads (8 wave32 waves). Each block owns 16 batch
// rows and runs the full 2048-step scan. Each wave owns a 32-column slice of
// the hidden state and keeps:
//   - W_hh (zero-diag) as 16 persistent f16 WMMA B-fragments (128 VGPRs)
//   - W_in as 2 persistent B-fragments
//   - h (f32, C-layout) and Cst (f32) tiles in registers
// h is mirrored in LDS as f16 (double buffered) in a layout where an A
// fragment is two ds_load_b128 per lane; A fragments are loaded on the fly
// inside the K loop to keep the wave under 256 VGPRs. One barrier per step.

typedef _Float16 half_t;
typedef __attribute__((ext_vector_type(16))) _Float16 v16h;
typedef __attribute__((ext_vector_type(8)))  _Float16 v8h;
typedef __attribute__((ext_vector_type(8)))  float    v8f;

#define T_STEPS 2048
#define BATCH   512
#define IN_F    28
#define NHID    256
#define NCLS    2

#define HSTR 264   // halves per row of h16 (padded: bank-friendly b128 loads)
#define XSTR 40    // halves per row of x16 (28 real + 4 zero-pad + 8 slack)

#define WMMA_F16(a, b, c) \
  __builtin_amdgcn_wmma_f32_16x16x32_f16(false, (a), false, (b), (short)0, (c), false, false)

// ---------------------------------------------------------------------------
// Kernel 1: fold all time-invariant additive terms into Cst[b][n] =
//           b_hh[n] + b_in[n] + b_ff[n] + sum_k hier[b,k] * W_ff[n,k]
// ---------------------------------------------------------------------------
__global__ void cst_kernel(const float* __restrict__ hier,
                           const float* __restrict__ W_ff,
                           const float* __restrict__ b_ff,
                           const float* __restrict__ b_hh,
                           const float* __restrict__ b_in,
                           float* __restrict__ Cst) {
  __shared__ float sh[NHID];
  const int b = blockIdx.x, n = threadIdx.x;
  sh[n] = hier[b * NHID + n];
  __syncthreads();
  float s = b_hh[n] + b_in[n] + b_ff[n];
  const float* wrow = W_ff + n * NHID;
#pragma unroll 8
  for (int k = 0; k < NHID; ++k) s += sh[k] * wrow[k];
  Cst[b * NHID + n] = s;
}

// ---------------------------------------------------------------------------
// Kernel 2: the fused persistent recurrence + readout.
// ---------------------------------------------------------------------------
__global__ __launch_bounds__(256) void rnn_kernel(
    const float* __restrict__ data,   // [T,B,I]
    const float* __restrict__ h0,     // [B,N]
    const float* __restrict__ W_in,   // [N,I]
    const float* __restrict__ W_hh,   // [N,N]
    const float* __restrict__ W_fc,   // [C,N]
    const float* __restrict__ b_fc,   // [C]
    const float* __restrict__ Cst,    // [B,N]
    float* __restrict__ out)          // [B,C]
{
  __shared__ half_t h16[2][16 * HSTR];   // f16 hidden state, double buffered
  __shared__ half_t x16[2][16 * XSTR];   // staged input tile, double buffered
  __shared__ float  hf[16 * NHID];       // final f32 h for readout

  const int tid  = threadIdx.x;
  const int wave = tid >> 5;
  const int lane = tid & 31;
  const int l    = lane & 15;   // row (A) / col (B,C) within 16-wide tile
  const int hi   = lane >> 4;   // lane-half select
  const int b0   = blockIdx.x * 16;   // batch rows owned by this block
  const int nb   = wave * 32;         // hidden columns owned by this wave

  // ---- one-time: pack W_hh (zero diagonal) into 16 persistent B fragments
  // B[k][n] = W_hh[n][k]; per-lane halves j: k = kc*32 + 8*hi + (j<8 ? j : j+8)
  v16h Bh[8][2];
#pragma unroll
  for (int kc = 0; kc < 8; ++kc) {
#pragma unroll
    for (int nt = 0; nt < 2; ++nt) {
      const int n = nb + nt * 16 + l;
      v16h f;
#pragma unroll
      for (int j = 0; j < 16; ++j) {
        const int k = kc * 32 + 8 * hi + ((j < 8) ? j : (j + 8));
        const float w = (k == n) ? 0.0f : W_hh[n * NHID + k];
        f[j] = (half_t)w;
      }
      Bh[kc][nt] = f;
    }
  }
  // ---- W_in B fragments (K padded 28 -> 32 with zeros)
  v16h Bx[2];
#pragma unroll
  for (int nt = 0; nt < 2; ++nt) {
    const int n = nb + nt * 16 + l;
    v16h f;
#pragma unroll
    for (int j = 0; j < 16; ++j) {
      const int k = 8 * hi + ((j < 8) ? j : (j + 8));
      const float w = (k < IN_F) ? W_in[n * IN_F + k] : 0.0f;
      f[j] = (half_t)w;
    }
    Bx[nt] = f;
  }
  // ---- h0 and Cst tiles in C/D layout: (lane,v) <-> (M = v + 8*hi, N = l)
  v8f hC0, hC1, cst0, cst1;
#pragma unroll
  for (int v = 0; v < 8; ++v) {
    const int m = v + 8 * hi;
    const int n = nb + l;
    hC0[v]  = h0 [(b0 + m) * NHID + n];
    hC1[v]  = h0 [(b0 + m) * NHID + n + 16];
    cst0[v] = Cst[(b0 + m) * NHID + n];
    cst1[v] = Cst[(b0 + m) * NHID + n + 16];
  }

  // ---- prologue: zero x pad columns, write h0 (f16) to buffer 0, stage t=0
  for (int e = tid; e < 2 * 16 * 12; e += 256) {
    const int buf = e / 192, r = e % 192;
    const int m = r / 12, i = 28 + (r % 12);
    x16[buf][m * XSTR + i] = (half_t)0.0f;
  }
  {
    const int n0 = nb + l;
#pragma unroll
    for (int v = 0; v < 8; ++v) {
      const int m = v + 8 * hi;
      h16[0][m * HSTR + n0]      = (half_t)hC0[v];
      h16[0][m * HSTR + n0 + 16] = (half_t)hC1[v];
    }
  }
  for (int e = tid; e < 16 * IN_F; e += 256) {
    const int m = e / IN_F, i = e - m * IN_F;
    x16[0][m * XSTR + i] = (half_t)data[(size_t)(b0 + m) * IN_F + i];
  }
  __syncthreads();

  // ---- the serial scan -----------------------------------------------------
#pragma unroll 1
  for (int t = 0; t < T_STEPS; ++t) {
    const int rb = t & 1, wb = rb ^ 1;

    // stage next timestep's input tile first: HBM latency overlaps the WMMAs
    if (t + 1 < T_STEPS) {
      for (int e = tid; e < 16 * IN_F; e += 256) {
        const int m = e / IN_F, i = e - m * IN_F;
        x16[wb][m * XSTR + i] =
            (half_t)data[((size_t)(t + 1) * BATCH + (b0 + m)) * IN_F + i];
      }
    }

    // A fragment of x(t)
    v16h Ax;
    {
      const half_t* xb = &x16[rb][l * XSTR + 8 * hi];
      v8h lo = *(const v8h*)(xb);
      v8h hh = *(const v8h*)(xb + 16);
      Ax = __builtin_shufflevector(lo, hh, 0,1,2,3,4,5,6,7,8,9,10,11,12,13,14,15);
    }

    // two interleaved accumulator chains (independent -> hides WMMA latency)
    v8f acc0 = cst0, acc1 = cst1;
    acc0 = WMMA_F16(Ax, Bx[0], acc0);
    acc1 = WMMA_F16(Ax, Bx[1], acc1);

    // h A-fragments loaded on the fly (keeps liveness < 256 VGPRs)
    const half_t* abase = &h16[rb][l * HSTR + 8 * hi];
#pragma unroll
    for (int kc = 0; kc < 8; ++kc) {
      v8h lo = *(const v8h*)(abase + kc * 32);
      v8h hh = *(const v8h*)(abase + kc * 32 + 16);
      v16h A = __builtin_shufflevector(lo, hh, 0,1,2,3,4,5,6,7,8,9,10,11,12,13,14,15);
      acc0 = WMMA_F16(A, Bh[kc][0], acc0);
      acc1 = WMMA_F16(A, Bh[kc][1], acc1);
    }

    // leaky-integrator update: h' = leaky_relu(0.5*(h + acc))
#pragma unroll
    for (int v = 0; v < 8; ++v) {
      const float p0 = 0.5f * (hC0[v] + acc0[v]);
      const float p1 = 0.5f * (hC1[v] + acc1[v]);
      hC0[v] = (p0 >= 0.0f) ? p0 : 0.01f * p0;
      hC1[v] = (p1 >= 0.0f) ? p1 : 0.01f * p1;
    }

    // publish h(t+1) as f16 into the other buffer
    {
      half_t* wp = &h16[wb][l + nb];
#pragma unroll
      for (int v = 0; v < 8; ++v) {
        const int m = v + 8 * hi;
        wp[m * HSTR]      = (half_t)hC0[v];
        wp[m * HSTR + 16] = (half_t)hC1[v];
      }
    }
    __syncthreads();
  }

  // ---- readout: out[b][c] = hT[b,:] . W_fc[c,:] + b_fc[c]  (f32 path)
  {
    const int n0 = nb + l;
#pragma unroll
    for (int v = 0; v < 8; ++v) {
      const int m = v + 8 * hi;
      hf[m * NHID + n0]      = hC0[v];
      hf[m * NHID + n0 + 16] = hC1[v];
    }
  }
  __syncthreads();
  if (tid < 32) {
    const int m = tid >> 1, c = tid & 1;
    float s = b_fc[c];
    const float* wrow = W_fc + c * NHID;
#pragma unroll 8
    for (int n = 0; n < NHID; ++n) s += hf[m * NHID + n] * wrow[n];
    out[(b0 + m) * NCLS + c] = s;
  }
}

// ---------------------------------------------------------------------------
extern "C" void kernel_launch(void* const* d_in, const int* in_sizes, int n_in,
                              void* d_out, int out_size, void* d_ws, size_t ws_size,
                              hipStream_t stream) {
  (void)in_sizes; (void)n_in; (void)out_size; (void)ws_size;
  const float* data = (const float*)d_in[0];   // [T,B,I]
  const float* hier = (const float*)d_in[1];   // [B,N]
  const float* h0   = (const float*)d_in[2];   // [B,N]
  const float* W_in = (const float*)d_in[3];   // [N,I]
  const float* b_in = (const float*)d_in[4];   // [N]
  const float* W_hh = (const float*)d_in[5];   // [N,N]
  const float* b_hh = (const float*)d_in[6];   // [N]
  const float* W_ff = (const float*)d_in[7];   // [N,N]
  const float* b_ff = (const float*)d_in[8];   // [N]
  const float* W_fc = (const float*)d_in[9];   // [C,N]
  const float* b_fc = (const float*)d_in[10];  // [C]
  float* out = (float*)d_out;
  float* Cst = (float*)d_ws;                   // [B,N] f32 = 512 KB scratch

  cst_kernel<<<BATCH, NHID, 0, stream>>>(hier, W_ff, b_ff, b_hh, b_in, Cst);
  rnn_kernel<<<BATCH / 16, 256, 0, stream>>>(data, h0, W_in, W_hh, W_fc, b_fc,
                                             Cst, out);
}
